// GeoAwarePooling_77841987273180
// MI455X (gfx1250) — matrix operand order repaired and checked by reading, not compiled, not run on hardware
//
#include <hip/hip_runtime.h>
#include <hip/hip_bf16.h>
#include <math.h>

// ---------------------------------------------------------------------------
// GeoAwarePooling for MI455X (gfx1250, wave32, WMMA f32_16x16x32_f16)
//
// View-invariance: local_xyz / global_xyz / weights are identical across the
// 4 views (xyz is broadcast), so the MLP runs once per (b, n).
//
// Pipeline:
//   K0 init      : gmax = -inf
//   K1 pass1     : per 128-pt tile: h=relu(LN(x@W1+b1)) [VALU],
//                  local=LN(h@W2+b2) [WMMA], atomicMax channel max -> gmax
//   K2 gb        : gb[b] = gmax[b] @ W3[C:2C] + b3  (tiny)
//   K3 pass2     : recompute local, h2=relu(LN(local@W3a + gb)) [WMMA],
//                  w = 2*sigmoid(h2@W4) -> wbuf
//   K4 pool      : out[b,v,c] = sum_n m*w*feat / (sum_n m*w + 1e-8) + gmax
//                  (one block per (b,c), prefetch-assisted stream)
// ---------------------------------------------------------------------------

typedef __attribute__((ext_vector_type(16))) _Float16 v16h;
typedef __attribute__((ext_vector_type(8)))  _Float16 v8h;
typedef __attribute__((ext_vector_type(8)))  float    v8f;

#define BB 4
#define NP 40000
#define CC 128
#define TILE 128          // points per workgroup tile
#define NTHREADS 256      // 8 waves; wave i owns row-tile i (16 points)
#define H_STRIDE 136      // f16 halves per row (pad vs 128 for LDS banks)
#define L_STRIDE 129      // f32 per row (pad -> conflict-free column walks)
#define LN_EPS 1e-5f
#define NTILES ((NP + TILE - 1) / TILE)

// ---- workspace layout (floats) --------------------------------------------
// gmax : [B*C]        offset 0
// gb   : [B*C]        offset 512
// wbuf : [B*N]        offset 1024
#define WS_GMAX 0
#define WS_GB   (BB * CC)
#define WS_WBUF (2 * BB * CC)

__device__ __forceinline__ void atomicMaxF(float* addr, float v) {
    if (v >= 0.0f) atomicMax((int*)addr, __float_as_int(v));
    else           atomicMin((unsigned int*)addr, __float_as_uint(v));
}

// Pre-swizzle a [128][128] row-major f32 weight matrix into B-fragment-linear
// f16 LDS: frag[((s*8 + j)*32 + lane)*16 + h], K(h) = 32*s + 8*(lane>>4) + (h<8 ? h : h+8),
// col = j*16 + (lane&15).  Each lane's 16 halves are contiguous (32B) so the
// per-fragment load is two ds_load_b128.
__device__ void fillWfrag(_Float16* frag, const float* W, int tid) {
    for (int f = tid; f < 32 * 32; f += NTHREADS) {
        int tile = f >> 5, lane = f & 31;
        int s = tile >> 3, j = tile & 7;
        int lo = lane & 15, hi = lane >> 4;
        int col = j * 16 + lo;
        _Float16* dst = frag + (tile * 32 + lane) * 16;
#pragma unroll
        for (int h = 0; h < 16; ++h) {
            int k = 32 * s + 8 * hi + (h < 8 ? h : h + 8);
            dst[h] = (_Float16)W[k * CC + col];
        }
    }
}

// A-fragment (16x32 f16) from row-major f16 LDS (stride H_STRIDE).
// ISA layout: lane lo holds row M=lo; halves 0..7 -> K = kBase+8*hi+{0..7},
// halves 8..15 -> K = kBase+16+8*hi+{0..7}.  Two contiguous 16B runs.
__device__ __forceinline__ v16h loadA(const _Float16* base, int rowBase,
                                      int kBase, int lane) {
    int lo = lane & 15, hi = lane >> 4;
    const _Float16* r = base + (rowBase + lo) * H_STRIDE + kBase + 8 * hi;
    v8h a0 = *(const v8h*)r;
    v8h a1 = *(const v8h*)(r + 16);
    v16h a;
#pragma unroll
    for (int i = 0; i < 8; ++i) { a[i] = a0[i]; a[i + 8] = a1[i]; }
    return a;
}

// One 128x128 GEMM over the tile: out[p][c] = (A @ Wfrag)[p][c] + bias[c].
// Wave `wave` owns rows 16*wave .. 16*wave+15; loops 8 col-tiles x 4 k-steps.
__device__ void gemm128(const _Float16* A, const _Float16* Wfrag,
                        float* outF32, const float* bias,
                        int wave, int lane) {
    int lo = lane & 15, hi = lane >> 4;
    int rowBase = wave * 16;
    for (int j = 0; j < 8; ++j) {
        v8f acc = {};
#pragma unroll
        for (int s = 0; s < 4; ++s) {
            v16h a = loadA(A, rowBase, 32 * s, lane);
            v16h bf = *(const v16h*)&Wfrag[((s * 8 + j) * 32 + lane) * 16];
            acc = __builtin_amdgcn_wmma_f32_16x16x32_f16(
                false, a, false, bf, (short)0, acc, false, false);
        }
        int col = j * 16 + lo;
        float bv = bias[col];
#pragma unroll
        for (int r = 0; r < 8; ++r) {
            // D layout: VGPR r holds row rowBase + r + 8*hi, col = lo
            outF32[(rowBase + r + 8 * hi) * L_STRIDE + col] = acc[r] + bv;
        }
    }
}

// ---- K0: init gmax to -inf -------------------------------------------------
__global__ void init_kernel(float* gmax) {
    for (int i = threadIdx.x; i < BB * CC; i += blockDim.x)
        gmax[i] = -INFINITY;
}

// ---- K1 / K3: main per-point MLP passes ------------------------------------
// prm layout: W1s[0..383] b1@384 g1@512 be1@640 b2@768 g2@896 be2@1024
//             g3@1152 be3@1280 gb@1408 W4@1536
template <bool PASS2>
__global__ __launch_bounds__(NTHREADS)
void pass_kernel(const float* __restrict__ xyz,
                 const float* __restrict__ W1, const float* __restrict__ b1,
                 const float* __restrict__ g1, const float* __restrict__ be1,
                 const float* __restrict__ W2, const float* __restrict__ b2,
                 const float* __restrict__ g2, const float* __restrict__ be2,
                 const float* __restrict__ W3,
                 const float* __restrict__ g3, const float* __restrict__ be3,
                 const float* __restrict__ W4,
                 float* __restrict__ gmax, const float* __restrict__ gb,
                 float* __restrict__ wbuf) {
    __shared__ _Float16 hA[TILE * H_STRIDE];      // 34,816 B (h / local f16)
    __shared__ float    locF[TILE * L_STRIDE];    // 66,048 B (GEMM staging)
    __shared__ _Float16 Wfrag[32 * 32 * 16];      // 32,768 B (W2, then W3a)
    __shared__ float    prm[384 + 10 * 128];      //  6,656 B

    const int tid  = threadIdx.x;
    const int lane = tid & 31;
    const int wave = tid >> 5;
    const int b    = blockIdx.y;
    const int n0   = blockIdx.x * TILE;
    const int nvalid = min(TILE, NP - n0);

    for (int i = tid; i < 384; i += NTHREADS) prm[i] = W1[i];
    for (int i = tid; i < CC; i += NTHREADS) {
        prm[384 + i] = b1[i];  prm[512 + i]  = g1[i];  prm[640 + i]  = be1[i];
        prm[768 + i] = b2[i];  prm[896 + i]  = g2[i];  prm[1024 + i] = be2[i];
        if (PASS2) {
            prm[1152 + i] = g3[i];        prm[1280 + i] = be3[i];
            prm[1408 + i] = gb[b * CC + i]; prm[1536 + i] = W4[i];
        }
    }
    fillWfrag(Wfrag, W2, tid);
    __syncthreads();

    // ---- stage 0: h = relu(LN(x @ W1 + b1))  (thread-per-point) ----
    if (tid < TILE) {
        float x0 = 0.f, x1 = 0.f, x2 = 0.f;
        if (tid < nvalid) {
            int n = n0 + tid;
            x0 = xyz[(b * 3 + 0) * NP + n];
            x1 = xyz[(b * 3 + 1) * NP + n];
            x2 = xyz[(b * 3 + 2) * NP + n];
        }
        float* yrow = &locF[tid * L_STRIDE];
        float s1 = 0.f, s2 = 0.f;
        for (int c = 0; c < CC; ++c) {
            float y = prm[384 + c];
            y = fmaf(x0, prm[c], y);
            y = fmaf(x1, prm[128 + c], y);
            y = fmaf(x2, prm[256 + c], y);
            yrow[c] = y; s1 += y; s2 += y * y;
        }
        float mean = s1 * (1.f / CC);
        float var  = s2 * (1.f / CC) - mean * mean;
        float rstd = rsqrtf(var + LN_EPS);
        _Float16* hrow = &hA[tid * H_STRIDE];
        for (int c = 0; c < CC; ++c) {
            float v = (yrow[c] - mean) * rstd * prm[512 + c] + prm[640 + c];
            hrow[c] = (_Float16)fmaxf(v, 0.f);
        }
    }
    __syncthreads();

    // ---- stage 1: locF = h @ W2 + b2  (WMMA) ----
    gemm128(hA, Wfrag, locF, &prm[768], wave, lane);
    __syncthreads();

    // ---- stage 2: local = LN(locF; g2, be2) -> hA (f16) ----
    if (tid < TILE) {
        const float* row = &locF[tid * L_STRIDE];
        float s1 = 0.f, s2 = 0.f;
        for (int c = 0; c < CC; ++c) { float v = row[c]; s1 += v; s2 += v * v; }
        float mean = s1 * (1.f / CC);
        float rstd = rsqrtf(s2 * (1.f / CC) - mean * mean + LN_EPS);
        _Float16* drow = &hA[tid * H_STRIDE];
        for (int c = 0; c < CC; ++c)
            drow[c] = (_Float16)((row[c] - mean) * rstd * prm[896 + c] + prm[1024 + c]);
    }
    if (PASS2) fillWfrag(Wfrag, W3, tid);   // W3 rows [0,C) = W3a
    __syncthreads();

    if (!PASS2) {
        // per-block channel max over valid points -> atomic max into gmax
        if (tid < CC) {
            float m = -INFINITY;
            for (int p = 0; p < nvalid; ++p)
                m = fmaxf(m, (float)hA[p * H_STRIDE + tid]);
            atomicMaxF(&gmax[b * CC + tid], m);
        }
    } else {
        // ---- stage 3: locF = local @ W3a + (global@W3b + b3)  (WMMA) ----
        gemm128(hA, Wfrag, locF, &prm[1408], wave, lane);
        __syncthreads();
        // ---- stage 4: w = 2*sigmoid( relu(LN(locF; g3,be3)) @ W4 ) ----
        if (tid < nvalid) {
            const float* row = &locF[tid * L_STRIDE];
            float s1 = 0.f, s2 = 0.f;
            for (int c = 0; c < CC; ++c) { float v = row[c]; s1 += v; s2 += v * v; }
            float mean = s1 * (1.f / CC);
            float rstd = rsqrtf(s2 * (1.f / CC) - mean * mean + LN_EPS);
            float dot = 0.f;
            for (int c = 0; c < CC; ++c) {
                float h2 = fmaxf((row[c] - mean) * rstd * prm[1152 + c] + prm[1280 + c], 0.f);
                dot = fmaf(h2, prm[1536 + c], dot);
            }
            wbuf[b * NP + n0 + tid] = 2.f / (1.f + expf(-dot));
        }
    }
}

// ---- K2: gb[b][j] = b3[j] + sum_c gmax[b][c] * W3[C + c][j] ---------------
__global__ void gb_kernel(const float* __restrict__ W3,
                          const float* __restrict__ b3,
                          const float* __restrict__ gmax,
                          float* __restrict__ gb) {
    int b = blockIdx.x, j = threadIdx.x;
    float s = b3[j];
    for (int c = 0; c < CC; ++c)
        s = fmaf(gmax[b * CC + c], W3[(CC + c) * CC + j], s);
    gb[b * CC + j] = s;
}

// ---- K4: fused pooling + finalize (one block per (b,c)) --------------------
__device__ __forceinline__ float blockReduce(float v, float* red, int t) {
    red[t] = v; __syncthreads();
    for (int s = 128; s > 0; s >>= 1) {
        if (t < s) red[t] += red[t + s];
        __syncthreads();
    }
    float r = red[0]; __syncthreads();
    return r;
}

__global__ __launch_bounds__(256)
void pool_kernel(const float* __restrict__ feat,
                 const float* __restrict__ masks,
                 const float* __restrict__ wbuf,
                 const float* __restrict__ gmax,
                 float* __restrict__ out) {
    __shared__ float red[256];
    const int c = blockIdx.x, b = blockIdx.y, t = threadIdx.x;
    const float* f  = feat + ((size_t)(b * CC + c)) * NP;
    const float* wb = wbuf + (size_t)b * NP;
    const float* m0 = masks + (size_t)b * 4 * NP;
    float a[4] = {0.f, 0.f, 0.f, 0.f};
    float dn[4] = {0.f, 0.f, 0.f, 0.f};
    for (int n = t; n < NP; n += 256) {
        __builtin_prefetch(f + n + 4096, 0, 1);   // global_prefetch_b8
        float fv = f[n];
        float wv = wb[n];
#pragma unroll
        for (int v = 0; v < 4; ++v) {
            float mw = m0[v * NP + n] * wv;
            a[v]  = fmaf(mw, fv, a[v]);
            dn[v] += mw;
        }
    }
    float ra[4], rd[4];
#pragma unroll
    for (int v = 0; v < 4; ++v) {
        ra[v] = blockReduce(a[v], red, t);
        rd[v] = blockReduce(dn[v], red, t);
    }
    if (t == 0) {
        float g = gmax[b * CC + c];
#pragma unroll
        for (int v = 0; v < 4; ++v)
            out[(b * 4 + v) * CC + c] = ra[v] / (rd[v] + 1e-8f) + g;
    }
}

// ---------------------------------------------------------------------------
extern "C" void kernel_launch(void* const* d_in, const int* in_sizes, int n_in,
                              void* d_out, int out_size, void* d_ws, size_t ws_size,
                              hipStream_t stream) {
    (void)in_sizes; (void)n_in; (void)out_size; (void)ws_size;
    const float* xyz   = (const float*)d_in[0];
    const float* feat  = (const float*)d_in[1];
    const float* masks = (const float*)d_in[2];
    const float* W1 = (const float*)d_in[3];  const float* b1  = (const float*)d_in[4];
    const float* g1 = (const float*)d_in[5];  const float* be1 = (const float*)d_in[6];
    const float* W2 = (const float*)d_in[7];  const float* b2  = (const float*)d_in[8];
    const float* g2 = (const float*)d_in[9];  const float* be2 = (const float*)d_in[10];
    const float* W3 = (const float*)d_in[11]; const float* b3  = (const float*)d_in[12];
    const float* g3 = (const float*)d_in[13]; const float* be3 = (const float*)d_in[14];
    const float* W4 = (const float*)d_in[15];

    float* out  = (float*)d_out;
    float* ws   = (float*)d_ws;
    float* gmax = ws + WS_GMAX;
    float* gb   = ws + WS_GB;
    float* wbuf = ws + WS_WBUF;

    dim3 grid(NTILES, BB);

    init_kernel<<<1, 256, 0, stream>>>(gmax);
    pass_kernel<false><<<grid, NTHREADS, 0, stream>>>(
        xyz, W1, b1, g1, be1, W2, b2, g2, be2, W3, g3, be3, W4,
        gmax, gb, wbuf);
    gb_kernel<<<BB, CC, 0, stream>>>(W3, b3, gmax, gb);
    pass_kernel<true><<<grid, NTHREADS, 0, stream>>>(
        xyz, W1, b1, g1, be1, W2, b2, g2, be2, W3, g3, be3, W4,
        gmax, gb, wbuf);
    pool_kernel<<<dim3(CC, BB), 256, 0, stream>>>(feat, masks, wbuf, gmax, out);
}
